// Kalman_filter_34041910788634
// MI455X (gfx1250) — matrix operand order repaired and checked by reading, not compiled, not run on hardware
//
#include <hip/hip_runtime.h>
#include <stddef.h>

// ---------------------------------------------------------------------------
// Problem constants (from reference)
// ---------------------------------------------------------------------------
#define N_DIM  2048
#define HIDDEN 4096
#define OUT_DIM 512
#define T_LEN  8

// GEMM tiling
#define BM 128
#define BN 128
#define BK 32
#define STAGES (N_DIM / BK)    // 64 K-stages
#define NBLK_Y (HIDDEN / BN)   // 32 N-blocks -> 32 deterministic partials/row

typedef __attribute__((ext_vector_type(16))) __bf16 v16bf;
typedef __attribute__((ext_vector_type(8)))  float  v8f;

// ---------------------------------------------------------------------------
// bf16 helpers (manual, round-to-nearest-even)
// ---------------------------------------------------------------------------
__device__ __forceinline__ unsigned short f32_to_bf16(float f) {
    unsigned int u = __float_as_uint(f);
    u += 0x7FFFu + ((u >> 16) & 1u);
    return (unsigned short)(u >> 16);
}
__device__ __forceinline__ float bf16_to_f32(unsigned short h) {
    return __uint_as_float(((unsigned int)h) << 16);
}

// Load a 16-element bf16 fragment from two contiguous 16-byte LDS chunks.
__device__ __forceinline__ v16bf ldfrag(const unsigned short* p0,
                                        const unsigned short* p1) {
    union { v16bf v; uint4 q[2]; } u;
    u.q[0] = *(const uint4*)p0;
    u.q[1] = *(const uint4*)p1;
    return u.v;
}

__device__ __forceinline__ v8f wmma_bf16(v16bf a, v16bf b, v8f c) {
    // (neg_a, A, neg_b, B, c_mod, C, reuse_a, reuse_b)
    return __builtin_amdgcn_wmma_f32_16x16x32_bf16(false, a, false, b,
                                                   (short)0, c, false, false);
}

// ---------------------------------------------------------------------------
// CDNA5 async global->LDS copy (ASYNCcnt-tracked).  GVS mode:
//   mem_addr = SGPR_base + VGPR_I32_offset ; LDS dest = wave-relative addr in
//   vdst VGPR (low 32 bits of a generic pointer to __shared__).
// ---------------------------------------------------------------------------
__device__ __forceinline__ void async_ld16(const void* base, unsigned goff,
                                           unsigned laddr) {
    asm volatile("global_load_async_to_lds_b128 %0, %1, %2"
                 :: "v"(laddr), "v"(goff), "s"(base)
                 : "memory");
}

// Issue one BK-deep stage: A(hi/lo) from Q rows m0..m0+127, B(hi/lo) from
// W1 rows n0..n0+127, K columns [kt, kt+BK).  8 async b128 per thread.
__device__ __forceinline__ void issue_stage(
        const unsigned short* Qhi, const unsigned short* Qlo,
        const unsigned short* W1hi, const unsigned short* W1lo,
        unsigned short* dAhi, unsigned short* dAlo,
        unsigned short* dBhi, unsigned short* dBlo,
        int m0, int n0, int kt, int tid) {
    #pragma unroll
    for (int i = 0; i < 2; ++i) {
        int c   = tid + 256 * i;      // 0..511
        int row = c >> 2;             // 0..127
        int col = (c & 3) * 8;        // 0,8,16,24 (elements)
        unsigned offA = (unsigned)(((m0 + row) * N_DIM + kt + col) * 2);
        unsigned offB = (unsigned)(((n0 + row) * N_DIM + kt + col) * 2);
        int l = row * BK + col;
        async_ld16(Qhi,  offA, (unsigned)(uintptr_t)&dAhi[l]);
        async_ld16(Qlo,  offA, (unsigned)(uintptr_t)&dAlo[l]);
        async_ld16(W1hi, offB, (unsigned)(uintptr_t)&dBhi[l]);
        async_ld16(W1lo, offB, (unsigned)(uintptr_t)&dBlo[l]);
    }
}

// ---------------------------------------------------------------------------
// k_split: fp32 -> (bf16 hi, bf16 lo) elementwise, grid-stride
// ---------------------------------------------------------------------------
__global__ __launch_bounds__(256) void k_split(const float* __restrict__ src,
                                               unsigned short* __restrict__ hi,
                                               unsigned short* __restrict__ lo,
                                               int n) {
    for (int i = blockIdx.x * 256 + threadIdx.x; i < n; i += gridDim.x * 256) {
        float x = src[i];
        unsigned short h = f32_to_bf16(x);
        hi[i] = h;
        lo[i] = f32_to_bf16(x - bf16_to_f32(h));
    }
}

// ---------------------------------------------------------------------------
// k_v: v[j] = sum_i W2[i][j] * e7[i]   (W2 is [N_DIM][HIDDEN] row-major)
// ---------------------------------------------------------------------------
__global__ __launch_bounds__(256) void k_v(const float* __restrict__ W2,
                                           const float* __restrict__ e7,
                                           float* __restrict__ v) {
    int j = blockIdx.x * 256 + threadIdx.x;   // 0..HIDDEN-1
    float acc = 0.f;
    for (int i = 0; i < N_DIM; ++i)
        acc += W2[(size_t)i * HIDDEN + j] * e7[i];
    v[j] = acc;
}

// ---------------------------------------------------------------------------
// k_rowdot: y[row] = (optional relu)( dot(A[row,:], x) + bias[row] )
// ---------------------------------------------------------------------------
__global__ __launch_bounds__(256) void k_rowdot(const float* __restrict__ A,
                                                const float* __restrict__ x,
                                                const float* __restrict__ bias,
                                                float* __restrict__ y,
                                                int K, int do_relu) {
    __shared__ float red[256];
    int row = blockIdx.x;
    const float* a = A + (size_t)row * K;
    float p = 0.f;
    for (int k = threadIdx.x; k < K; k += 256) p += a[k] * x[k];
    red[threadIdx.x] = p;
    __syncthreads();
    for (int s = 128; s > 0; s >>= 1) {
        if (threadIdx.x < s) red[threadIdx.x] += red[threadIdx.x + s];
        __syncthreads();
    }
    if (threadIdx.x == 0) {
        float r = red[0] + (bias ? bias[row] : 0.f);
        if (do_relu) r = fmaxf(r, 0.f);
        y[row] = r;
    }
}

// ---------------------------------------------------------------------------
// k_scalar: s_all = dot(w + o0, T7) + dot(b2, e7)
// ---------------------------------------------------------------------------
__global__ __launch_bounds__(256) void k_scalar(const float* __restrict__ w,
                                                const float* __restrict__ o0,
                                                const float* __restrict__ T7,
                                                const float* __restrict__ b2,
                                                const float* __restrict__ e7,
                                                float* __restrict__ s_all) {
    __shared__ float red[256];
    float p = 0.f;
    for (int i = threadIdx.x; i < N_DIM; i += 256)
        p += (w[i] + o0[i]) * T7[i] + b2[i] * e7[i];
    red[threadIdx.x] = p;
    __syncthreads();
    for (int s = 128; s > 0; s >>= 1) {
        if (threadIdx.x < s) red[threadIdx.x] += red[threadIdx.x + s];
        __syncthreads();
    }
    if (threadIdx.x == 0) s_all[0] = red[0];
}

// ---------------------------------------------------------------------------
// k_gemm_hv: flagship fused WMMA kernel, async double-buffered.
//   P[m][n] = sum_k Q[m][k] * W1[n][k]   (split-bf16: 3 WMMAs per K-step)
//   epilogue: rowpart[m] = sum_n relu(P[m][n] + b1[n]) * v[n]
//   writes hv_part[blockIdx.y][m]  (deterministic, no atomics)
// ---------------------------------------------------------------------------
__global__ __launch_bounds__(256) void k_gemm_hv(
        const unsigned short* __restrict__ Qhi,
        const unsigned short* __restrict__ Qlo,
        const unsigned short* __restrict__ W1hi,
        const unsigned short* __restrict__ W1lo,
        const float* __restrict__ b1,
        const float* __restrict__ v,
        float* __restrict__ hv_part) {
    __shared__ unsigned short sAhi[2][BM * BK];   // 8KB each buffer
    __shared__ unsigned short sAlo[2][BM * BK];
    __shared__ unsigned short sBhi[2][BN * BK];
    __shared__ unsigned short sBlo[2][BN * BK];   // total 64KB

    const int tid  = threadIdx.x;
    const int wave = tid >> 5;        // 0..7 : owns rows [wave*16, wave*16+16)
    const int lane = tid & 31;
    const int half = lane >> 4;       // 0 or 1
    const int l15  = lane & 15;
    const int m0 = blockIdx.x * BM;
    const int n0 = blockIdx.y * BN;

    v8f acc[8];
    #pragma unroll
    for (int ns = 0; ns < 8; ++ns)
        #pragma unroll
        for (int r = 0; r < 8; ++r) acc[ns][r] = 0.f;

    // Prologue: prefetch stage 0 into buffer 0.
    issue_stage(Qhi, Qlo, W1hi, W1lo,
                sAhi[0], sAlo[0], sBhi[0], sBlo[0], m0, n0, 0, tid);

    for (int s = 0; s < STAGES; ++s) {
        if (s + 1 < STAGES) {
            int nb = (s + 1) & 1;
            issue_stage(Qhi, Qlo, W1hi, W1lo,
                        sAhi[nb], sAlo[nb], sBhi[nb], sBlo[nb],
                        m0, n0, (s + 1) * BK, tid);
            // Async loads complete in order: <=8 outstanding means the 8
            // loads of stage s have landed (only stage s+1 still in flight).
            asm volatile("s_wait_asynccnt 0x8" ::: "memory");
        } else {
            asm volatile("s_wait_asynccnt 0x0" ::: "memory");
        }
        __syncthreads();

        const int cb = s & 1;
        // A fragment: lane<16 holds K {0..7,16..23}; lane>=16 holds
        // K {8..15,24..31} of row (wave*16 + l15).  (ISA 05_wmma 7.12.2)
        const int arow = (wave * 16 + l15) * BK + half * 8;
        v16bf ahi = ldfrag(&sAhi[cb][arow], &sAhi[cb][arow + 16]);
        v16bf alo = ldfrag(&sAlo[cb][arow], &sAlo[cb][arow + 16]);
        #pragma unroll
        for (int ns = 0; ns < 8; ++ns) {
            // B fragment: lane<16 -> K {0..7,16..23} of col N=l15; lane>=16 ->
            // K {8..15,24..31}.  B[k][n] = W1[n0+n][k]
            const int brow = (ns * 16 + l15) * BK + half * 8;
            v16bf bhi = ldfrag(&sBhi[cb][brow], &sBhi[cb][brow + 16]);
            v16bf blo = ldfrag(&sBlo[cb][brow], &sBlo[cb][brow + 16]);
            acc[ns] = wmma_bf16(ahi, bhi, acc[ns]);
            acc[ns] = wmma_bf16(ahi, blo, acc[ns]);
            acc[ns] = wmma_bf16(alo, bhi, acc[ns]);
        }
        // All waves must finish reading buf[s&1] before anyone overwrites it
        // with stage s+2's async loads in the next iteration.
        __syncthreads();
    }

    // Epilogue: relu(P + b1) * v, reduced over this block's 128 columns.
    // C/D layout: VGPR r -> (M = r + 8*half, N = l15) within the 16x16 tile.
    float rsum[8];
    #pragma unroll
    for (int r = 0; r < 8; ++r) rsum[r] = 0.f;
    #pragma unroll
    for (int ns = 0; ns < 8; ++ns) {
        int nc = n0 + ns * 16 + l15;
        float bias = b1[nc];
        float vf   = v[nc];
        #pragma unroll
        for (int r = 0; r < 8; ++r) {
            float val = fmaxf(acc[ns][r] + bias, 0.f);
            rsum[r] += val * vf;
        }
    }
    // Reduce across each 16-lane half (same M rows live in 16 lanes).
    #pragma unroll
    for (int r = 0; r < 8; ++r) {
        float x = rsum[r];
        x += __shfl_xor(x, 1, 32);
        x += __shfl_xor(x, 2, 32);
        x += __shfl_xor(x, 4, 32);
        x += __shfl_xor(x, 8, 32);
        rsum[r] = x;
    }
    if (l15 == 0) {
        float* dst = hv_part + (size_t)blockIdx.y * N_DIM
                   + m0 + wave * 16 + half * 8;
        #pragma unroll
        for (int r = 0; r < 8; ++r) dst[r] = rsum[r];
    }
}

// ---------------------------------------------------------------------------
// k_combine: last[i] = s_all + qe[i] + sum_b hv_part[b][i]   (deterministic)
// ---------------------------------------------------------------------------
__global__ __launch_bounds__(256) void k_combine(const float* __restrict__ qe,
                                                 const float* __restrict__ hv_part,
                                                 const float* __restrict__ s_all,
                                                 float* __restrict__ last) {
    int i = blockIdx.x * 256 + threadIdx.x;
    if (i >= N_DIM) return;
    float a = s_all[0] + qe[i];
    #pragma unroll
    for (int b = 0; b < NBLK_Y; ++b) a += hv_part[(size_t)b * N_DIM + i];
    last[i] = a;
}

// ---------------------------------------------------------------------------
// k_out: out[o] = dot(fc_w[o,:], last) + fc_b[o]
// ---------------------------------------------------------------------------
__global__ __launch_bounds__(256) void k_out(const float* __restrict__ fcw,
                                             const float* __restrict__ fcb,
                                             const float* __restrict__ last,
                                             float* __restrict__ out) {
    __shared__ float red[256];
    int o = blockIdx.x;
    const float* a = fcw + (size_t)o * N_DIM;
    float p = 0.f;
    for (int i = threadIdx.x; i < N_DIM; i += 256) p += a[i] * last[i];
    red[threadIdx.x] = p;
    __syncthreads();
    for (int s = 128; s > 0; s >>= 1) {
        if (threadIdx.x < s) red[threadIdx.x] += red[threadIdx.x + s];
        __syncthreads();
    }
    if (threadIdx.x == 0) out[o] = red[0] + fcb[o];
}

// ---------------------------------------------------------------------------
// kernel_launch
// ---------------------------------------------------------------------------
extern "C" void kernel_launch(void* const* d_in, const int* in_sizes, int n_in,
                              void* d_out, int out_size, void* d_ws, size_t ws_size,
                              hipStream_t stream) {
    const float* T   = (const float*)d_in[0];
    const float* e   = (const float*)d_in[1];
    const float* w   = (const float*)d_in[2];
    const float* Q   = (const float*)d_in[3];
    const float* W1  = (const float*)d_in[4];
    const float* b1  = (const float*)d_in[5];
    const float* W2  = (const float*)d_in[6];
    const float* b2  = (const float*)d_in[7];
    const float* fcw = (const float*)d_in[8];
    const float* fcb = (const float*)d_in[9];
    float* out = (float*)d_out;

    const float* T7 = T + (size_t)(T_LEN - 1) * N_DIM;
    const float* e7 = e + (size_t)(T_LEN - 1) * N_DIM;

    // Workspace carve-up (~50.6 MB)
    char* p = (char*)d_ws;
    unsigned short* Qhi  = (unsigned short*)p; p += (size_t)N_DIM * N_DIM * 2;
    unsigned short* Qlo  = (unsigned short*)p; p += (size_t)N_DIM * N_DIM * 2;
    unsigned short* W1hi = (unsigned short*)p; p += (size_t)HIDDEN * N_DIM * 2;
    unsigned short* W1lo = (unsigned short*)p; p += (size_t)HIDDEN * N_DIM * 2;
    float* v       = (float*)p; p += HIDDEN * sizeof(float);
    float* h0      = (float*)p; p += HIDDEN * sizeof(float);
    float* qe      = (float*)p; p += N_DIM * sizeof(float);
    float* o0      = (float*)p; p += N_DIM * sizeof(float);
    float* hv_part = (float*)p; p += (size_t)NBLK_Y * N_DIM * sizeof(float);
    float* last    = (float*)p; p += N_DIM * sizeof(float);
    float* s_all   = (float*)p; p += 256;

    // 1) fp32 -> bf16 hi/lo splits of Q and W1
    k_split<<<2048, 256, 0, stream>>>(Q,  Qhi,  Qlo,  N_DIM * N_DIM);
    k_split<<<4096, 256, 0, stream>>>(W1, W1hi, W1lo, HIDDEN * N_DIM);

    // 2) small matvecs
    k_v<<<HIDDEN / 256, 256, 0, stream>>>(W2, e7, v);
    k_rowdot<<<HIDDEN, 256, 0, stream>>>(W1, w, b1, h0, N_DIM, 1);     // h0
    k_rowdot<<<N_DIM, 256, 0, stream>>>(Q, e7, nullptr, qe, N_DIM, 0); // qe

    // 3) flagship fused WMMA GEMM (async double-buffered staging)
    dim3 grid(N_DIM / BM, HIDDEN / BN);   // 16 x 32 = 512 workgroups
    k_gemm_hv<<<grid, 256, 0, stream>>>(Qhi, Qlo, W1hi, W1lo, b1, v, hv_part);

    // 4) scalars and output
    k_rowdot<<<N_DIM, 256, 0, stream>>>(W2, h0, b2, o0, HIDDEN, 0);    // o0
    k_scalar<<<1, 256, 0, stream>>>(w, o0, T7, b2, e7, s_all);
    k_combine<<<(N_DIM + 255) / 256, 256, 0, stream>>>(qe, hv_part, s_all, last);
    k_out<<<OUT_DIM, 256, 0, stream>>>(fcw, fcb, last, out);
}
